// CollaborativeRNNModel_64673617543719
// MI455X (gfx1250) — matrix operand order repaired and checked by reading, not compiled, not run on hardware
//
#include <hip/hip_runtime.h>
#include <hip/hip_bf16.h>

// ---------------------------------------------------------------------------
// CollaborativeRNN (GRU-ish) for MI455X / gfx1250:
//   wave32 + WMMA bf16 + async-to-LDS (ASYNCcnt) + TDM (TENSORcnt).
//   B=32, T=256, H=1024, V=10001
// ---------------------------------------------------------------------------

typedef __attribute__((ext_vector_type(16))) __bf16 v16bf;
typedef __attribute__((ext_vector_type(8)))  __bf16 v8bf;
typedef __attribute__((ext_vector_type(8)))  float  v8f;
typedef __attribute__((ext_vector_type(4)))  unsigned v4u;
typedef __attribute__((ext_vector_type(8)))  unsigned v8u;

#define B_   32
#define T_   256
#define H_   1024
#define V_   10001
#define VP_  10016           // V padded to 16
#define NTO_ 626             // VP_/16 N-tiles for W_out
#define NBLK 32              // persistent grid size (must be co-resident)

// ---------------- CDNA5 async-to-LDS primitives -----------------------------
// VDST = LDS byte address (low 32 bits of the generic shared pointer: the
// LDS aperture maps flat->LDS by truncating to addr[31:0]).  VADDR = 64-bit
// global address VGPR pair, SADDR = off.  Tracked with ASYNCcnt.
__device__ inline void async_copy_b128(const void* gsrc, void* lds_dst) {
  unsigned lds_addr = (unsigned)(size_t)lds_dst;
  asm volatile("global_load_async_to_lds_b128 %0, %1, off"
               :: "v"(lds_addr), "v"(gsrc)
               : "memory");
}

template <int N>
__device__ inline void wait_asynccnt() {
  asm volatile("s_wait_asynccnt %0" :: "i"(N) : "memory");
}

// ---------------- CDNA5 Tensor Data Mover (TDM) -----------------------------
// One 2D descriptor DMA of a 32x1024 bf16 (64 KB) tile into LDS, with LDS
// padding of 16 B per 1024 B stored (pad_interval=256 dw, pad_amount=4 dw),
// so the effective LDS row stride is 2080 B = 1040 elements.
// D# per ISA ch.8: group0 = {flags, lds_addr, global_addr, type=2},
// group1 = {mask/data_size/pad, dims/strides}; groups 2/3 zero (dims unused).
#define LDS_LD2 1040
__device__ inline void tdm_load_32x1024(const __bf16* gsrc, __bf16* sm) {
  unsigned lds_addr = (unsigned)(size_t)sm;
  unsigned long long ga = (unsigned long long)(size_t)gsrc;
  v4u g0;
  g0.x = 1u;                                            // count=1, user D#
  g0.y = lds_addr;                                      // lds_addr[31:0]
  g0.z = (unsigned)(ga & 0xFFFFFFFFu);                  // global_addr[31:0]
  g0.w = (unsigned)((ga >> 32) & 0x1FFFFFFu) | (2u << 30); // addr[56:32]|type=2
  v8u g1;
  g1.s0 = (3u << 16)    // data_size = 3 (8-byte units)
        | (1u << 20)    // pad_enable
        | (7u << 22)    // pad_interval: 256 dwords (1024 B)
        | (3u << 25);   // pad_amount: 4 dwords (16 B)
  g1.s1 = 256u << 16;   // tensor_dim0 = 256 (8B units per 2048 B row)
  g1.s2 = 32u << 16;    // tensor_dim1 = 32 rows
  g1.s3 = 256u << 16;   // tile_dim0 = 256
  g1.s4 = 32u;          // tile_dim1 = 32, tile_dim2 = 0
  g1.s5 = 256u;         // tensor_dim0_stride = 256 (contiguous)
  g1.s6 = 0u;
  g1.s7 = 0u;
  v4u gz; gz.x = 0u; gz.y = 0u; gz.z = 0u; gz.w = 0u;
  asm volatile("tensor_load_to_lds %0, %1, %2, %3"
               :: "s"(g0), "s"(g1), "s"(gz), "s"(gz)
               : "memory");
}

// ---------------- WMMA fragment loaders (per CDNA5 ISA §7.12.2) -------------
// A (16x32 bf16): lane m=lane&15 holds row m; half=lane>>4 selects K-phase.
// elements 0..7  = K kcol0 + half*8 .. +7
// elements 8..15 = K kcol0 + 16 + half*8 .. +7
__device__ inline v16bf load_a_frag(const __bf16* sm, int row0, int ld,
                                    int kcol0, int lane) {
  int m = lane & 15, half = lane >> 4;
  const __bf16* base = sm + (size_t)(row0 + m) * ld + kcol0 + half * 8;
  union { v16bf v; v8bf h[2]; } u;
  u.h[0] = *(const v8bf*)(base);
  u.h[1] = *(const v8bf*)(base + 16);
  return u.v;
}

// A-frag loader for the TDM-padded recurrence staging: row stride 1040
// elements, plus the mid-row +8-element pad bump at column 512 (chunks are
// 8-element aligned so they never straddle the 512 boundary).
__device__ inline v16bf load_a_frag_tdm(const __bf16* sm, int row0,
                                        int kcol0, int lane) {
  int m = lane & 15, half = lane >> 4;
  int c0 = kcol0 + half * 8;
  int c1 = c0 + 16;
  const __bf16* rowp = sm + (size_t)(row0 + m) * LDS_LD2;
  union { v16bf v; v8bf h[2]; } u;
  u.h[0] = *(const v8bf*)(rowp + c0 + ((c0 >> 9) << 3));
  u.h[1] = *(const v8bf*)(rowp + c1 + ((c1 >> 9) << 3));
  return u.v;
}

// B (32x16 bf16) pre-packed: frag is 512 bf16; lane holds 16 contiguous
// elements at frag + lane*16 (element e <-> K=(lane>>4)*8+(e&7)+(e>>3)*16, N=lane&15).
__device__ inline v16bf load_b_frag(const __bf16* frag, int lane) {
  const __bf16* base = frag + lane * 16;
  union { v16bf v; v8bf h[2]; } u;
  u.h[0] = *(const v8bf*)(base);
  u.h[1] = *(const v8bf*)(base + 8);
  return u.v;
}

__device__ inline float sigmoidf_(float x) { return 1.0f / (1.0f + __expf(-x)); }

// ---------------- pack fp32 weights -> bf16 B-fragment layout ---------------
__global__ void pack_bf16_frags(const float* __restrict__ src,
                                __bf16* __restrict__ dst,
                                int NT, int Nsrc, int ld, int total) {
  int tid = blockIdx.x * blockDim.x + threadIdx.x;
  if (tid >= total) return;
  int e    = tid & 15;
  int lane = (tid >> 4) & 31;
  int frag = tid >> 9;
  int nt = frag % NT;
  int kt = frag / NT;
  int k = kt * 32 + ((lane >> 4) << 3) + (e & 7) + ((e >> 3) << 4);   // K within [0,1024)
  int n = nt * 16 + (lane & 15);
  float v = (n < Nsrc) ? src[(size_t)k * ld + n] : 0.0f;
  dst[tid] = (__bf16)v;
}

__global__ void init_h_kernel(const float* __restrict__ h0,
                              float* __restrict__ h_f32,
                              __bf16* __restrict__ hb) {
  int i = blockIdx.x * blockDim.x + threadIdx.x;
  if (i < B_ * H_) { float v = h0[i]; h_f32[i] = v; hb[i] = (__bf16)v; }
}

// ---------------- grid-wide barrier (monotone phase counter) ----------------
__device__ inline void grid_barrier(unsigned* cnt, unsigned target) {
  __syncthreads();
  if (threadIdx.x == 0) {
    __threadfence();
    __hip_atomic_fetch_add(cnt, 1u, __ATOMIC_RELEASE, __HIP_MEMORY_SCOPE_AGENT);
    while (__hip_atomic_load(cnt, __ATOMIC_ACQUIRE, __HIP_MEMORY_SCOPE_AGENT) < target)
      __builtin_amdgcn_s_sleep(1);
  }
  __syncthreads();
}

// ---------------- persistent recurrence kernel ------------------------------
__global__ void __launch_bounds__(256)
recurrent_kernel(const int* __restrict__ items,
                 const float* __restrict__ E_ru, const float* __restrict__ b_ru,
                 const float* __restrict__ E_c,  const float* __restrict__ b_c,
                 const __bf16* __restrict__ Wru_pk,
                 const __bf16* __restrict__ Wc_pk,
                 float* __restrict__ h_f32, __bf16* __restrict__ hb,
                 __bf16* __restrict__ rh, float* __restrict__ u_buf,
                 float* __restrict__ ru_lin, float* __restrict__ c_lin,
                 __bf16* __restrict__ states, unsigned* __restrict__ cnt,
                 float* __restrict__ h_out) {
  __shared__ __bf16 smh[32 * LDS_LD2];  // 66,560 B staging for h / r*h

  const int tid  = threadIdx.x;
  const int lane = tid & 31;
  const int wv   = tid >> 5;                 // wave32: 8 waves / block
  const int gw   = blockIdx.x * 8 + wv;      // 0..255 global wave id
  const int gtid = blockIdx.x * 256 + tid;   // 0..8191
  const int m    = lane & 15, half = lane >> 4;

  unsigned target = 0;

  for (int t = 0; t < T_; ++t) {
    // ---- GEMM1: ru_lin = h @ W_ru  (32x1024 x 1024x2048) ----
    if (wv == 0) {                            // one TDM DMA per block
      tdm_load_32x1024(hb, smh);
      __builtin_amdgcn_s_wait_tensorcnt(0);
    }
    __syncthreads();
    if (gw < 256) {                           // 2 M-tiles x 128 N-tiles
      int mt = gw & 1, nt = gw >> 1;
      v8f acc = {};
      for (int kt = 0; kt < 32; ++kt) {
        v16bf a = load_a_frag_tdm(smh, mt * 16, kt * 32, lane);
        const __bf16* bb = Wru_pk + ((size_t)(kt * 128 + nt)) * 512;
        if (kt < 31) __builtin_prefetch(bb + 128 * 512, 0, 1);
        v16bf b = load_b_frag(bb, lane);
        acc = __builtin_amdgcn_wmma_f32_16x16x32_bf16(
            false, a, false, b, (short)0, acc, false, false);
      }
      float* out = ru_lin + (size_t)(mt * 16 + half * 8) * 2048 + nt * 16 + m;
#pragma unroll
      for (int g = 0; g < 8; ++g) out[(size_t)g * 2048] = acc[g];
    }
    target += NBLK; grid_barrier(cnt, target);

    // ---- elementwise 1: r,u = sigmoid(x_ru + ru_lin + b_ru); rh = r*h ----
    for (int i = gtid; i < B_ * H_; i += NBLK * 256) {
      int b = i >> 10, j = i & 1023;
      int it = items[b * T_ + t];
      float xr = E_ru[(size_t)it * 2048 + j];
      float xu = E_ru[(size_t)it * 2048 + 1024 + j];
      float r = sigmoidf_(xr + ru_lin[(size_t)b * 2048 + j] + b_ru[j]);
      float u = sigmoidf_(xu + ru_lin[(size_t)b * 2048 + 1024 + j] + b_ru[1024 + j]);
      u_buf[i] = u;
      rh[i] = (__bf16)(r * h_f32[i]);
    }
    target += NBLK; grid_barrier(cnt, target);

    // ---- GEMM2: c_lin = (r*h) @ W_c  (32x1024 x 1024x1024) ----
    if (wv == 0) {
      tdm_load_32x1024(rh, smh);
      __builtin_amdgcn_s_wait_tensorcnt(0);
    }
    __syncthreads();
    if (gw < 128) {                           // 2 M-tiles x 64 N-tiles
      int mt = gw & 1, nt = gw >> 1;
      v8f acc = {};
      for (int kt = 0; kt < 32; ++kt) {
        v16bf a = load_a_frag_tdm(smh, mt * 16, kt * 32, lane);
        const __bf16* bb = Wc_pk + ((size_t)(kt * 64 + nt)) * 512;
        if (kt < 31) __builtin_prefetch(bb + 64 * 512, 0, 1);
        v16bf b = load_b_frag(bb, lane);
        acc = __builtin_amdgcn_wmma_f32_16x16x32_bf16(
            false, a, false, b, (short)0, acc, false, false);
      }
      float* out = c_lin + (size_t)(mt * 16 + half * 8) * 1024 + nt * 16 + m;
#pragma unroll
      for (int g = 0; g < 8; ++g) out[(size_t)g * 1024] = acc[g];
    }
    target += NBLK; grid_barrier(cnt, target);

    // ---- elementwise 2: c = tanh(..); h = u*h + (1-u)*c; emit state ----
    for (int i = gtid; i < B_ * H_; i += NBLK * 256) {
      int b = i >> 10, j = i & 1023;
      int it = items[b * T_ + t];
      float xc = E_c[(size_t)it * 1024 + j];
      float c = tanhf(xc + c_lin[i] + b_c[j]);
      float u = u_buf[i];
      float hn = u * h_f32[i] + (1.0f - u) * c;
      h_f32[i] = hn;
      hb[i] = (__bf16)hn;
      states[((size_t)b * T_ + t) * H_ + j] = (__bf16)hn;
    }
    target += NBLK; grid_barrier(cnt, target);
  }

  // h_final -> d_out[0 : B*H]
  for (int i = gtid; i < B_ * H_; i += NBLK * 256) h_out[i] = h_f32[i];
}

// ---------------- output GEMM: logits = states @ W_out ----------------------
// block tile 128(M) x 64(N), 8 waves, one 16-row strip / wave, 4 N-frags.
// A and B tiles double-buffered in LDS via async-to-LDS; each thread issues
// exactly 3 async b128s per k-step, so s_wait_asynccnt 3 keeps one buffer in
// flight (async loads complete in order per ISA §10.7/§4).
#define ALD 40   // LDS stride for A tile (80 B: 16B-aligned, conflict-free)
__global__ void __launch_bounds__(256)
logits_gemm(const __bf16* __restrict__ states,
            const __bf16* __restrict__ Wout_pk,
            float* __restrict__ logits) {
  __shared__ __bf16 As[2][128 * ALD];   // 2 x 10,240 B
  __shared__ __bf16 Bs[2][4 * 512];     // 2 x  4,096 B
  const int nb = blockIdx.x;                  // 0..156 (N blocks of 64)
  const int mb = blockIdx.y;                  // 0..63  (M blocks of 128)
  const int tid = threadIdx.x;
  const int wave = tid >> 5, lane = tid & 31;
  const int m = lane & 15, half = lane >> 4;

  // per-thread issue of one k-step's tiles (2 A-transfers + 1 B-transfer)
  auto issue_tile = [&](int kt, int buf) {
    int idx = tid;
#pragma unroll
    for (int r = 0; r < 2; ++r, idx += 256) {      // A: 128 rows x 4 chunks
      int row = idx >> 2, hc = idx & 3;
      async_copy_b128(states + (size_t)(mb * 128 + row) * H_ + kt * 32 + hc * 8,
                      &As[buf][row * ALD + hc * 8]);
    }
    int f = tid >> 6, chunk = tid & 63;            // B: 4 frags x 64 chunks
    int ntg = nb * 4 + f;
    if (ntg >= NTO_) ntg = NTO_ - 1;               // clamp (pads unused accs)
    async_copy_b128(Wout_pk + ((size_t)kt * NTO_ + ntg) * 512 + chunk * 8,
                    &Bs[buf][f * 512 + chunk * 8]);
  };

  v8f acc[4] = {{}, {}, {}, {}};

  issue_tile(0, 0);
  for (int kt = 0; kt < 32; ++kt) {
    int cur = kt & 1;
    if (kt + 1 < 32) {
      issue_tile(kt + 1, cur ^ 1);
      wait_asynccnt<3>();      // retire current buffer's 3, keep next in flight
    } else {
      wait_asynccnt<0>();
    }
    __syncthreads();

    v16bf a = load_a_frag(As[cur], wave * 16, ALD, 0, lane);
#pragma unroll
    for (int f = 0; f < 4; ++f) {
      if (nb * 4 + f < NTO_) {
        v16bf b = load_b_frag(&Bs[cur][f * 512], lane);
        acc[f] = __builtin_amdgcn_wmma_f32_16x16x32_bf16(
            false, a, false, b, (short)0, acc[f], false, false);
      }
    }
    __syncthreads();           // protect buf reuse on next iteration
  }

#pragma unroll
  for (int f = 0; f < 4; ++f) {
    int col = nb * 64 + f * 16 + m;
    if (col < V_) {
#pragma unroll
      for (int g = 0; g < 8; ++g) {
        int row = mb * 128 + wave * 16 + half * 8 + g;
        logits[(size_t)row * V_ + col] = acc[f][g];
      }
    }
  }
}

// ---------------------------------------------------------------------------
extern "C" void kernel_launch(void* const* d_in, const int* in_sizes, int n_in,
                              void* d_out, int out_size, void* d_ws, size_t ws_size,
                              hipStream_t stream) {
  const int*   items = (const int*)  d_in[0];
  const float* h0    = (const float*)d_in[1];
  const float* E_ru  = (const float*)d_in[2];
  const float* W_ru  = (const float*)d_in[3];
  const float* b_ru  = (const float*)d_in[4];
  const float* E_c   = (const float*)d_in[5];
  const float* W_c   = (const float*)d_in[6];
  const float* b_c   = (const float*)d_in[7];
  const float* W_out = (const float*)d_in[8];

  float* h_out  = (float*)d_out;             // [B*H]
  float* logits = (float*)d_out + B_ * H_;   // [B*T, V]

  // workspace carve-up (256 B aligned)
  char* p = (char*)d_ws;
  auto take = [&](size_t bytes) { char* r = p; p += (bytes + 255) & ~size_t(255); return r; };
  unsigned* cnt     = (unsigned*)take(256);
  __bf16*  Wru_pk   = (__bf16*)take((size_t)32 * 128  * 512 * 2);
  __bf16*  Wc_pk    = (__bf16*)take((size_t)32 * 64   * 512 * 2);
  __bf16*  Wout_pk  = (__bf16*)take((size_t)32 * NTO_ * 512 * 2);
  __bf16*  states   = (__bf16*)take((size_t)B_ * T_ * H_ * 2);
  float*   h_f32    = (float*) take((size_t)B_ * H_ * 4);
  float*   u_buf    = (float*) take((size_t)B_ * H_ * 4);
  float*   ru_lin   = (float*) take((size_t)B_ * 2 * H_ * 4);
  float*   c_lin    = (float*) take((size_t)B_ * H_ * 4);
  __bf16*  hb       = (__bf16*)take((size_t)B_ * H_ * 2);
  __bf16*  rh       = (__bf16*)take((size_t)B_ * H_ * 2);

  hipMemsetAsync(cnt, 0, 256, stream);       // barrier phase counter

  // pack weights to bf16 fragment layouts
  {
    int tot = 32 * 128 * 512;
    pack_bf16_frags<<<(tot + 255) / 256, 256, 0, stream>>>(W_ru, Wru_pk, 128, 2 * H_, 2 * H_, tot);
  }
  {
    int tot = 32 * 64 * 512;
    pack_bf16_frags<<<(tot + 255) / 256, 256, 0, stream>>>(W_c, Wc_pk, 64, H_, H_, tot);
  }
  {
    int tot = 32 * NTO_ * 512;
    pack_bf16_frags<<<(tot + 255) / 256, 256, 0, stream>>>(W_out, Wout_pk, NTO_, V_, V_, tot);
  }
  init_h_kernel<<<(B_ * H_ + 255) / 256, 256, 0, stream>>>(h0, h_f32, hb);

  // persistent GRU recurrence (grid-wide barriers; NBLK blocks co-resident)
  recurrent_kernel<<<NBLK, 256, 0, stream>>>(
      items, E_ru, b_ru, E_c, b_c, Wru_pk, Wc_pk,
      h_f32, hb, rh, u_buf, ru_lin, c_lin, states, cnt, h_out);

  // logits = states @ W_out
  dim3 grid((NTO_ + 3) / 4, (B_ * T_) / 128);
  logits_gemm<<<grid, 256, 0, stream>>>(states, Wout_pk, logits);
}
